// GATComm_43001212567750
// MI455X (gfx1250) — matrix operand order repaired and checked by reading, not compiled, not run on hardware
//
#include <hip/hip_runtime.h>
#include <hip/hip_bf16.h>

// ---------------------------------------------------------------------------
// GCN forward on gfx1250 (MI455X): 3x (A_norm @ (H @ W) + b) with ELU.
// One workgroup (8 wave32s) per batch graph. GEMMs via v_wmma_f32_16x16x32_bf16
// with register-resident A-fragments (one M-tile per wave). Feature tile is
// DMA'd into LDS by the Tensor Data Mover overlapped with adjacency work.
// f32->bf16 conversion via single-instruction v_fma_mixlo/hi_bf16 (VOP3P);
// WMMA->inline-asm RAW hazard covered by explicit 4x v_nop (ISA 7.12.1).
// ---------------------------------------------------------------------------

#define Bsz   1024
#define Nn    128
#define FIN   128
#define FHID  256
#define FOUT  128

// LDS strides (bf16 elements), padded to avoid 64-bank conflicts (256B rows)
#define LDA   136   // A_norm  [128][136]
#define LDH   264   // H       [128][264]  (holds up to 256 features)
#define LDT   136   // T = (H@W)^T  [256][136]

#define OFF_AS   0
#define OFF_H    34816                   // 128*136*2
#define OFF_T    102400                  // + 128*264*2  (also x staging, 64KB)
#define OFF_DINV 172032                  // + 256*136*2
#define SMEM_BYTES 172544                // + 128*4

typedef __attribute__((ext_vector_type(16))) __bf16 v16bf;
typedef __attribute__((ext_vector_type(8)))  float  v8f;
typedef __attribute__((ext_vector_type(4)))  float  f4v;
typedef __attribute__((ext_vector_type(4)))  unsigned uint4v;
typedef __attribute__((ext_vector_type(2)))  unsigned uint2v;
typedef __attribute__((ext_vector_type(4)))  unsigned tdm_g0_t;
typedef __attribute__((ext_vector_type(8)))  int      tdm_g1_t;
typedef __attribute__((ext_vector_type(4)))  int      tdm_g2_t;

#define ZERO8 {0.f, 0.f, 0.f, 0.f, 0.f, 0.f, 0.f, 0.f}

// ---- f32 -> bf16 via VOP3P V_FMA_MIXLO/MIXHI_BF16 (ISA 15.10 op 62/63) ----
static __device__ __forceinline__ unsigned pack2(float lo, float hi) {
    unsigned r;
    asm("v_fma_mixlo_bf16 %0, %1, 1.0, 0\n\t"
        "v_fma_mixhi_bf16 %0, %2, 1.0, 0"
        : "=&v"(r) : "v"(lo), "v"(hi));
    return r;
}
static __device__ __forceinline__ unsigned short f2bf(float v) {
    unsigned r;
    asm("v_fma_mixlo_bf16 %0, %1, 1.0, 0" : "=&v"(r) : "v"(v));
    return (unsigned short)r;
}

// Convert a whole WMMA accumulator tile (8xf32 -> 4x packed bf16x2) in one
// asm block. Leading 4x v_nop satisfies the WMMA->VALU RAW hazard (4 coexec
// NOPs for *bf16 WMMA, ISA 7.12.1) since the hazard recognizer does not pad
// ahead of inline asm; data deps on all 8 floats pin it after the WMMA.
static __device__ __forceinline__ uint4v pack8_wmma(v8f a) {
    unsigned r0, r1, r2, r3;
    asm("v_nop\n\tv_nop\n\tv_nop\n\tv_nop\n\t"
        "v_fma_mixlo_bf16 %0, %4, 1.0, 0\n\t"
        "v_fma_mixhi_bf16 %0, %5, 1.0, 0\n\t"
        "v_fma_mixlo_bf16 %1, %6, 1.0, 0\n\t"
        "v_fma_mixhi_bf16 %1, %7, 1.0, 0\n\t"
        "v_fma_mixlo_bf16 %2, %8, 1.0, 0\n\t"
        "v_fma_mixhi_bf16 %2, %9, 1.0, 0\n\t"
        "v_fma_mixlo_bf16 %3, %10, 1.0, 0\n\t"
        "v_fma_mixhi_bf16 %3, %11, 1.0, 0"
        : "=&v"(r0), "=&v"(r1), "=&v"(r2), "=&v"(r3)
        : "v"(a[0]), "v"(a[1]), "v"(a[2]), "v"(a[3]),
          "v"(a[4]), "v"(a[5]), "v"(a[6]), "v"(a[7]));
    uint4v r = { r0, r1, r2, r3 };
    return r;
}

// A-fragment (16x32 bf16, MxK) from a row-major [M][ld] bf16 buffer.
// Contiguous k (same as load_frag_b): lanes 0-15 -> M=lane, K=0..15; lanes 16-31 -> K=16..31.
static __device__ __forceinline__ v16bf load_frag_a(const unsigned short* base,
                                                    int ld, int mBase, int kBase,
                                                    int lane) {
    union { unsigned u[8]; v16bf v; } f;
    const int row = mBase + (lane & 15);
    const int ko  = (lane & 16) ? 16 : 0;
    const unsigned short* p = base + row * ld + kBase + ko;
#pragma unroll
    for (int j = 0; j < 4; ++j) f.u[j]     = *(const unsigned*)(p + 2 * j);
#pragma unroll
    for (int j = 0; j < 4; ++j) f.u[4 + j] = *(const unsigned*)(p + 8 + 2 * j);
    return f.v;
}

// B-fragment (32x16 bf16, KxN) from K-contiguous [N][ld] bf16 storage.
// Contiguous k: lane -> N=lane&15; lanes 0-15 hold K=0..15, lanes 16-31 K=16..31.
static __device__ __forceinline__ v16bf load_frag_b(const unsigned short* base,
                                                    int ld, int nBase, int kBase,
                                                    int lane) {
    union { unsigned u[8]; v16bf v; } f;
    const int col = nBase + (lane & 15);
    const unsigned short* p = base + col * ld + kBase + ((lane & 16) ? 16 : 0);
#pragma unroll
    for (int j = 0; j < 8; ++j) f.u[j] = *(const unsigned*)(p + 2 * j);
    return f.v;
}

// One GCN layer: T = (H @ W)^T  then  dst = A_norm @ T + b  (+ELU).
// Each wave owns M-tile mi = wid*16; A-side fragments live in registers.
template <int K, int FO, bool ELUACT, bool FINAL>
static __device__ __forceinline__ void gcn_layer(unsigned short* H,
                                                 unsigned short* T,
                                                 const v16bf sF[4],
                                                 const unsigned short* WtL,
                                                 const float* bias,
                                                 float* outB, int tid) {
    const int wid   = tid >> 5;
    const int lane  = tid & 31;
    const int mi    = wid * 16;
    const int col16 = lane & 15;
    const int hi8   = (lane & 16) ? 8 : 0;

    // ---- Step A: T[FO][N] = (H[N][K] @ W[K][FO])^T ----
    v16bf aF[K / 32];
#pragma unroll
    for (int k = 0; k < K / 32; ++k) aF[k] = load_frag_a(H, LDH, mi, 32 * k, lane);

    for (int ni = 0; ni < FO; ni += 16) {
        v8f acc = ZERO8;
#pragma unroll
        for (int k = 0; k < K / 32; ++k) {
            v16bf w = load_frag_b(WtL, K, ni, 32 * k, lane);
            acc = __builtin_amdgcn_wmma_f32_16x16x32_bf16(
                false, aF[k], false, w, (short)0, acc, false, false);
        }
        // D layout: lane -> col = ni+col16; VGPR j -> row mi+hi8+j.
        unsigned short* dst = T + (ni + col16) * LDT + mi + hi8;   // 16B aligned
        *(uint4v*)dst = pack8_wmma(acc);
    }
    __syncthreads();

    // ---- Step B: result = A_norm @ T^T + bias (A_norm frags resident) ----
    for (int ni = 0; ni < FO; ni += 16) {
        v8f acc = ZERO8;
#pragma unroll
        for (int k = 0; k < Nn / 32; ++k) {
            v16bf bm = load_frag_b(T, LDT, ni, 32 * k, lane);
            acc = __builtin_amdgcn_wmma_f32_16x16x32_bf16(
                false, sF[k], false, bm, (short)0, acc, false, false);
        }
        const int col = ni + col16;
        const float bv = bias[col];
        const int r0 = mi + hi8;
        // First acc reader is a compiler-visible v_add_f32 -> hazard NOPs
        // are inserted automatically by the backend here.
#pragma unroll
        for (int j = 0; j < 8; ++j) {
            float v = acc[j] + bv;
            if (ELUACT) v = (v > 0.f) ? v : (__expf(v) - 1.f);
            if (FINAL)  outB[(r0 + j) * FO + col] = v;
            else        H[(r0 + j) * LDH + col]   = f2bf(v);
        }
    }
    __syncthreads();
}

// Pre-transpose & convert weights to bf16 in workspace:
//  W1t[256][128] | W2t[256][256] | W3t[128][256]   (256 KB total)
__global__ void gcn_prep_weights(const float* __restrict__ W1,
                                 const float* __restrict__ W2,
                                 const float* __restrict__ W3,
                                 unsigned short* __restrict__ wt) {
    const int tid    = blockIdx.x * blockDim.x + threadIdx.x;
    const int stride = gridDim.x * blockDim.x;
    for (int i = tid; i < FHID * FIN; i += stride) {        // W1t[o][k]
        int o = i >> 7, k = i & 127;
        wt[i] = f2bf(W1[k * FHID + o]);
    }
    unsigned short* w2t = wt + FHID * FIN;
    for (int i = tid; i < FHID * FHID; i += stride) {       // W2t[o][k]
        int o = i >> 8, k = i & 255;
        w2t[i] = f2bf(W2[k * FHID + o]);
    }
    unsigned short* w3t = wt + FHID * FIN + FHID * FHID;
    for (int i = tid; i < FOUT * FHID; i += stride) {       // W3t[o][k]
        int o = i >> 8, k = i & 255;
        w3t[i] = f2bf(W3[k * FOUT + o]);
    }
}

__global__ __launch_bounds__(256)
void gcn_forward_kernel(const float* __restrict__ x,
                        const float* __restrict__ adj,
                        const float* __restrict__ b1,
                        const float* __restrict__ b2,
                        const float* __restrict__ b3,
                        const unsigned short* __restrict__ Wt,
                        float* __restrict__ out) {
    extern __shared__ char smem[];
    unsigned short* As   = (unsigned short*)(smem + OFF_AS);
    unsigned short* H    = (unsigned short*)(smem + OFF_H);
    unsigned short* T    = (unsigned short*)(smem + OFF_T);
    float*          dinv = (float*)(smem + OFF_DINV);

    const int tid  = threadIdx.x;
    const int wid  = tid >> 5;
    const int lane = tid & 31;
    const int b    = blockIdx.x;

    const float* adjB = adj + (size_t)b * Nn * Nn;
    const float* xB   = x   + (size_t)b * Nn * FIN;
    float*       outB = out + (size_t)b * Nn * FOUT;

    // ---- Kick off TDM: DMA x[b] (128x128 f32, 64KB) into LDS @ OFF_T,
    //      overlapped with the adjacency-normalization phase below. ----
#if __has_builtin(__builtin_amdgcn_tensor_load_to_lds)
    if (tid < 32) {   // one wave issues the descriptor (EXEC-independent op)
        unsigned long long ga = (unsigned long long)(uintptr_t)xB;
        tdm_g0_t g0 = { 1u,                               // count=1
                        (unsigned)OFF_T,                  // lds_addr
                        (unsigned)(ga & 0xFFFFFFFFu),     // global_addr lo
                        (unsigned)((ga >> 32) & 0x1FFFFFFu) | 0x80000000u }; // type=2
        // 1-D copy: data_size=4B, tensor_dim0=tile_dim0=16384, dim1=1
        tdm_g1_t g1 = { 0x20000,              // data_size=2 (4B)
                        0x40000000,           // tensor_dim0[15:0]=16384 << 16
                        0x10000,              // tensor_dim1=1
                        0x40000000,           // tile_dim0=16384 << 16
                        0,                    // tile_dim1/2 unused
                        16384,                // tensor_dim0_stride lo
                        0, 0 };
        tdm_g2_t z4 = { 0, 0, 0, 0 };
#if defined(__clang_major__) && __clang_major__ >= 23
        tdm_g1_t z8 = { 0, 0, 0, 0, 0, 0, 0, 0 };
        __builtin_amdgcn_tensor_load_to_lds(g0, g1, z4, z4, z8, 0);
#else
        __builtin_amdgcn_tensor_load_to_lds(g0, g1, z4, z4, 0);
#endif
    }
#endif

    // Pull the (L2-resident) weight block toward this WGP.
    for (int off = tid * 64; off < FHID * FIN + FHID * FHID + FOUT * FHID;
         off += 256 * 64)
        __builtin_prefetch(Wt + off, 0, 0);

    // ---- Phase 1a: degrees (with self loops) -> dinv ----
    for (int i = wid; i < Nn; i += 8) {
        float s = 0.f;
        for (int j = lane; j < Nn; j += 32) {
            float a = adjB[i * Nn + j];
            s += (a != 0.f || j == i) ? 1.f : 0.f;
        }
#pragma unroll
        for (int off = 16; off; off >>= 1) s += __shfl_xor(s, off, 32);
        if (lane == 0) dinv[i] = (s > 0.f) ? rsqrtf(s) : 0.f;
    }
    __syncthreads();

    // ---- Phase 1b: A_norm (bf16) into LDS, vectorized 4-wide ----
    for (int idx = tid; idx < (Nn * Nn) / 4; idx += 256) {
        int i = idx >> 5;            // row
        int j = (idx & 31) * 4;      // col
        f4v a = *(const f4v*)(adjB + i * Nn + j);
        float di = dinv[i];
        float v0 = ((a.x != 0.f || i == j + 0) ? 1.f : 0.f) * di * dinv[j + 0];
        float v1 = ((a.y != 0.f || i == j + 1) ? 1.f : 0.f) * di * dinv[j + 1];
        float v2 = ((a.z != 0.f || i == j + 2) ? 1.f : 0.f) * di * dinv[j + 2];
        float v3 = ((a.w != 0.f || i == j + 3) ? 1.f : 0.f) * di * dinv[j + 3];
        uint2v p = { pack2(v0, v1), pack2(v2, v3) };
        *(uint2v*)(As + i * LDA + j) = p;
    }

    // ---- Phase 1c: wait for TDM, convert x f32 -> H bf16 ----
#if __has_builtin(__builtin_amdgcn_tensor_load_to_lds)
    if (tid < 32) __builtin_amdgcn_s_wait_tensorcnt(0);
    __syncthreads();
    const float* xsrc = (const float*)(smem + OFF_T);
#else
    __syncthreads();
    const float* xsrc = xB;
#endif
    for (int idx = tid; idx < (Nn * FIN) / 4; idx += 256) {
        int i = idx >> 5;
        int j = (idx & 31) * 4;
        f4v a = *(const f4v*)(xsrc + i * FIN + j);
        uint2v p = { pack2(a.x, a.y), pack2(a.z, a.w) };
        *(uint2v*)(H + i * LDH + j) = p;
    }
    __syncthreads();

    // ---- A_norm fragments: loaded once, reused by all three layers ----
    v16bf sF[Nn / 32];
#pragma unroll
    for (int k = 0; k < Nn / 32; ++k)
        sF[k] = load_frag_a(As, LDA, wid * 16, 32 * k, lane);

    // ---- Phase 2: three WMMA layers ----
    const unsigned short* W1t = Wt;
    const unsigned short* W2t = Wt + FHID * FIN;
    const unsigned short* W3t = Wt + FHID * FIN + FHID * FHID;

    gcn_layer<FIN,  FHID, true,  false>(H, T, sF, W1t, b1, outB, tid);
    gcn_layer<FHID, FHID, true,  false>(H, T, sF, W2t, b2, outB, tid);
    gcn_layer<FHID, FOUT, false, true >(H, T, sF, W3t, b3, outB, tid);
}

extern "C" void kernel_launch(void* const* d_in, const int* in_sizes, int n_in,
                              void* d_out, int out_size, void* d_ws, size_t ws_size,
                              hipStream_t stream) {
    const float* x   = (const float*)d_in[0];
    const float* adj = (const float*)d_in[1];
    const float* W1  = (const float*)d_in[2];
    const float* b1  = (const float*)d_in[3];
    const float* W2  = (const float*)d_in[4];
    const float* b2  = (const float*)d_in[5];
    const float* W3  = (const float*)d_in[6];
    const float* b3  = (const float*)d_in[7];
    float* out       = (float*)d_out;
    unsigned short* wt = (unsigned short*)d_ws;   // 256 KB of bf16 weights

    gcn_prep_weights<<<256, 256, 0, stream>>>(W1, W2, W3, wt);
    gcn_forward_kernel<<<Bsz, 256, SMEM_BYTES, stream>>>(x, adj, b1, b2, b3, wt, out);
}
